// EquiRNN_6133213299290
// MI455X (gfx1250) — matrix operand order repaired and checked by reading, not compiled, not run on hardware
//
#include <hip/hip_runtime.h>
#include <hip/hip_bf16.h>
#include <math.h>

typedef __attribute__((ext_vector_type(16))) __bf16 v16bf;
typedef __attribute__((ext_vector_type(8)))  __bf16 v8bf;
typedef __attribute__((ext_vector_type(8)))  float  v8f;

constexpr int Gn = 24;        // groups
constexpr int Kn = 256;       // hidden dim
constexpr int Vn = 1024;      // vocab
constexpr int Tn = 256;       // time steps
constexpr int Bn = 2;         // batch
constexpr int KKn = Kn * Kn;                     // 65536 per weight matrix
constexpr long TBGK = (long)Tn * Bn * Gn * Kn;   // 3,145,728

// ---- WMMA fragment helpers (layouts per CDNA5 ISA 7.12.2) ----------------

// A-matrix 16x32 bf16: lane m (<16): row m, K = {k0..k0+7, k0+16..k0+23};
// lane m+16: row m, K = {k0+8..k0+15, k0+24..k0+31}.
static __device__ inline v16bf load_a_bf16(const __bf16* row, int k0, int half) {
  v8bf lo = *(const v8bf*)(row + k0 + half * 8);
  v8bf hi = *(const v8bf*)(row + k0 + 16 + half * 8);
  return __builtin_shufflevector(lo, hi, 0,1,2,3,4,5,6,7,8,9,10,11,12,13,14,15);
}

// B-matrix 32x16 bf16 from row-major W[K][256]: lane L holds column l0+L%16,
// K = {k0+8h..k0+8h+7, k0+16+8h..k0+23+8h}, h = L/16 (same order as A).
static __device__ inline v16bf load_b_bf16(const __bf16* w, int k0, int l0, int lane) {
  const __bf16* p = w + (size_t)(k0 + (lane >> 4) * 8) * Kn + l0 + (lane & 15);
  v16bf v;
#pragma unroll
  for (int s = 0; s < 16; ++s) v[s] = p[(size_t)((s & 7) + (s >> 3) * 16) * Kn];
  return v;
}

static __device__ inline v8f wmma_bf16(v16bf a, v16bf b, v8f c) {
  return __builtin_amdgcn_wmma_f32_16x16x32_bf16(
      /*neg_a=*/false, a, /*neg_b=*/false, b,
      /*c_mod=*/(short)0, c, /*reuse_a=*/false, /*reuse_b=*/false);
}

// ---- Prep kernels --------------------------------------------------------

__global__ void k_prep_misc(const float* __restrict__ bw, const float* __restrict__ bh,
                            const int* __restrict__ gidx, const float* __restrict__ h0,
                            float* __restrict__ bias, int* __restrict__ inv,
                            __bf16* __restrict__ s0) {
  int tid = blockIdx.x * blockDim.x + threadIdx.x;
  if (tid < Kn) bias[tid] = bw[tid] + bh[tid];
  if (tid < Gn * Gn) {
    int h = tid / Gn, g = tid % Gn;
    inv[h * Gn + gidx[h * Gn + g]] = g;   // inverse permutation per output group
  }
  if (tid < Bn * Gn * Kn) s0[tid] = (__bf16)h0[tid];
}

__global__ void k_cvt_w(const float* __restrict__ Ww, const float* __restrict__ Wh,
                        __bf16* __restrict__ oW, __bf16* __restrict__ oH) {
  long i = (long)blockIdx.x * blockDim.x + threadIdx.x;
  const long N = (long)Gn * KKn;
  if (i < N)          oW[i] = (__bf16)Ww[i];
  else if (i < 2 * N) oH[i - N] = (__bf16)Wh[i - N];
}

// Xg[t][b][g][k] = bf16( embedding[ inv_perm[g, tokens[b,t]] , k ] )
__global__ void k_gather(const int* __restrict__ tokens, const int* __restrict__ inv_perm,
                         const float* __restrict__ emb, __bf16* __restrict__ Xg) {
  long i = (long)blockIdx.x * blockDim.x + threadIdx.x;  // < TBGK
  int k = (int)(i & (Kn - 1));
  long rem = i >> 8;
  int g = (int)(rem % Gn); rem /= Gn;
  int b = (int)(rem & 1);
  int t = (int)(rem >> 1);
  int tok = tokens[b * Tn + t];
  int p = inv_perm[g * Vn + tok];
  Xg[i] = (__bf16)emb[(size_t)p * Kn + k];
}

// ---- Phase 1: time-parallel x-contribution -------------------------------
// PRE[(t*2+b)*G + h][l] = bias[l] + sum_g Xg[t,b,g,:] @ Ww[gidx[h,g]][:,l]
// Grid: 24 h * 32 rowtiles = 768 blocks; 256 threads = 8 waves; each wave
// owns two 16-wide column tiles (A-fragment reuse). Fragments are explicitly
// double-buffered so loads for iteration i+1 fly under iteration i's WMMAs.
__global__ void __launch_bounds__(256)
k_gemm_x(const __bf16* __restrict__ Xg, const __bf16* __restrict__ Ww,
         const int* __restrict__ gidx, const float* __restrict__ bias,
         float* __restrict__ PRE) {
  int h = blockIdx.x / 32;
  int rowtile = blockIdx.x % 32;          // rows r = (t,b), r = t*2+b
  int lane = threadIdx.x & 31;
  int wave = threadIdx.x >> 5;
  int half = lane >> 4, m = lane & 15;
  int l0 = wave * 32;
  int r = rowtile * 16 + m;
  const __bf16* arow_base = Xg + (size_t)r * (Gn * Kn);
  const int* gixh = gidx + h * Gn;

  v8f acc0 = {};
  v8f acc1 = {};

  const __bf16* w_cur = Ww + (size_t)gixh[0] * KKn;
  v16bf a_c  = load_a_bf16(arow_base, 0, half);
  v16bf b0_c = load_b_bf16(w_cur, 0, l0, lane);
  v16bf b1_c = load_b_bf16(w_cur, 0, l0 + 16, lane);

  for (int g = 0; g < Gn; ++g) {
    const __bf16* a_base  = arow_base + g * Kn;
    const __bf16* w_nxt   = (g + 1 < Gn) ? (Ww + (size_t)gixh[g + 1] * KKn) : w_cur;
    const __bf16* a_nbase = (g + 1 < Gn) ? (a_base + Kn) : a_base;
    // warm the WGP$ for the next weight matrix slice while crunching this one
    __builtin_prefetch(w_nxt + (size_t)lane * Kn + l0, 0, 1);
#pragma unroll
    for (int kc = 0; kc < 8; ++kc) {
      v16bf a_n, b0_n, b1_n;
      if (kc < 7) {
        int k0 = (kc + 1) * 32;
        a_n  = load_a_bf16(a_base, k0, half);
        b0_n = load_b_bf16(w_cur, k0, l0, lane);
        b1_n = load_b_bf16(w_cur, k0, l0 + 16, lane);
      } else {
        a_n  = load_a_bf16(a_nbase, 0, half);
        b0_n = load_b_bf16(w_nxt, 0, l0, lane);
        b1_n = load_b_bf16(w_nxt, 0, l0 + 16, lane);
      }
      acc0 = wmma_bf16(a_c, b0_c, acc0);
      acc1 = wmma_bf16(a_c, b1_c, acc1);
      a_c = a_n; b0_c = b0_n; b1_c = b1_n;
    }
    w_cur = w_nxt;
  }

#pragma unroll
  for (int i = 0; i < 8; ++i) {
    int R = rowtile * 16 + i + half * 8;        // R = t*2+b
    int c0 = l0 + m, c1 = l0 + 16 + m;
    size_t base = ((size_t)R * Gn + h) * Kn;
    PRE[base + c0] = acc0[i] + bias[c0];
    PRE[base + c1] = acc1[i] + bias[c1];
  }
}

// ---- Phase 2: recurrent step (launched once per t) -----------------------
// j-decomposition: pre[b,h] += sum_j state[b, inv_h(j)] @ Wh[j] — all 16
// (h,b)-rows of a tile share the same B matrix Wh[j].
// Grid: 3 rowtiles * 16 coltiles = 48 blocks of 128 threads (4 waves).
// Wave w reduces j = w, w+4, ..., w+20 (48 WMMAs, pipelined), partials are
// combined in LDS; all threads run the tanh epilogue.
__global__ void __launch_bounds__(128)
k_step(const float* __restrict__ PRE, const __bf16* __restrict__ Wh,
       const int* __restrict__ inv, const __bf16* __restrict__ s_in,
       __bf16* __restrict__ s_out, float* __restrict__ hidden,
       float* __restrict__ hT, int t) {
  __shared__ float red[4 * 256];

  int rowtile = blockIdx.x / 16;               // 0..2, rows r = h*2+b
  int l0 = (blockIdx.x % 16) * 16;
  int lane = threadIdx.x & 31;
  int wv = threadIdx.x >> 5;                   // 0..3
  int half = lane >> 4, m = lane & 15;
  int r = rowtile * 16 + m;
  const int* invh = inv + (r >> 1) * Gn;       // per-row h = r>>1
  const __bf16* sb = s_in + (size_t)(r & 1) * Gn * Kn;   // per-row b = r&1

  v8f acc = {};
  const __bf16* w_cur = Wh + (size_t)wv * KKn;
  v16bf a_c = load_a_bf16(sb + invh[wv] * Kn, 0, half);
  v16bf b_c = load_b_bf16(w_cur, 0, l0, lane);

  for (int jj = wv; jj < Gn; jj += 4) {        // 6 uniform iterations per wave
    const __bf16* a_base  = sb + invh[jj] * Kn;
    int jn = jj + 4;
    const __bf16* w_nxt   = (jn < Gn) ? (Wh + (size_t)jn * KKn) : w_cur;
    const __bf16* a_nbase = (jn < Gn) ? (sb + invh[jn] * Kn) : a_base;
    __builtin_prefetch(w_nxt + (size_t)lane * Kn + l0, 0, 1);
#pragma unroll
    for (int kc = 0; kc < 8; ++kc) {
      v16bf a_n, b_n;
      if (kc < 7) {
        int k0 = (kc + 1) * 32;
        a_n = load_a_bf16(a_base, k0, half);
        b_n = load_b_bf16(w_cur, k0, l0, lane);
      } else {
        a_n = load_a_bf16(a_nbase, 0, half);
        b_n = load_b_bf16(w_nxt, 0, l0, lane);
      }
      acc = wmma_bf16(a_c, b_c, acc);
      a_c = a_n; b_c = b_n;
    }
    w_cur = w_nxt;
  }

  // deposit this wave's 16x16 partial tile into LDS
#pragma unroll
  for (int i = 0; i < 8; ++i)
    red[wv * 256 + (i + half * 8) * 16 + m] = acc[i];
  __syncthreads();

  // cross-wave reduction + tanh epilogue: 2 tile elements per thread
  for (int e = threadIdx.x; e < 256; e += 128) {
    float s = red[e] + red[256 + e] + red[512 + e] + red[768 + e];
    int Ri = e >> 4;                 // row within tile
    int c  = e & 15;                 // col within tile
    int R  = rowtile * 16 + Ri;      // R = h*2+b
    int h  = R >> 1, b = R & 1;
    int col = l0 + c;
    size_t off = ((size_t)(t * Bn + b) * Gn + h) * Kn + col;  // [t][b][h][l]
    float v = tanhf(s + PRE[off]);
    hidden[off] = v;                            // hidden_all output
    size_t soff = ((size_t)b * Gn + h) * Kn + col;
    s_out[soff] = (__bf16)v;                    // next-step state (bf16)
    if (hT) hT[soff] = v;                       // h_T on last step
  }
}

// ---- Host launcher -------------------------------------------------------

extern "C" void kernel_launch(void* const* d_in, const int* in_sizes, int n_in,
                              void* d_out, int out_size, void* d_ws, size_t ws_size,
                              hipStream_t stream) {
  const int*   tokens   = (const int*)  d_in[0];
  const int*   inv_perm = (const int*)  d_in[1];
  const int*   gidx     = (const int*)  d_in[2];
  const float* emb      = (const float*)d_in[3];
  const float* Ww       = (const float*)d_in[4];
  const float* bw       = (const float*)d_in[5];
  const float* Wh       = (const float*)d_in[6];
  const float* bh       = (const float*)d_in[7];
  const float* h0       = (const float*)d_in[8];

  char* ws = (char*)d_ws;
  size_t off = 0;
  auto carve = [&](size_t bytes) {
    size_t r = off;
    off = (off + bytes + 255) & ~(size_t)255;
    return r;
  };
  float*  bias = (float*) (ws + carve((size_t)Kn * 4));
  __bf16* bWw  = (__bf16*)(ws + carve((size_t)Gn * KKn * 2));
  __bf16* bWh  = (__bf16*)(ws + carve((size_t)Gn * KKn * 2));
  __bf16* Xg   = (__bf16*)(ws + carve((size_t)TBGK * 2));
  float*  PRE  = (float*) (ws + carve((size_t)TBGK * 4));
  __bf16* s0   = (__bf16*)(ws + carve((size_t)Bn * Gn * Kn * 2));
  __bf16* s1   = (__bf16*)(ws + carve((size_t)Bn * Gn * Kn * 2));
  int*    inv  = (int*)   (ws + carve((size_t)Gn * Gn * 4));

  float* hidden = (float*)d_out;
  float* hT     = hidden + TBGK;

  k_prep_misc<<<(Bn * Gn * Kn + 255) / 256, 256, 0, stream>>>(bw, bh, gidx, h0,
                                                              bias, inv, s0);
  k_cvt_w<<<(int)((2L * Gn * KKn + 255) / 256), 256, 0, stream>>>(Ww, Wh, bWw, bWh);
  k_gather<<<(int)((TBGK + 255) / 256), 256, 0, stream>>>(tokens, inv_perm, emb, Xg);
  k_gemm_x<<<Gn * 32, 256, 0, stream>>>(Xg, bWw, gidx, bias, PRE);

  for (int t = 0; t < Tn; ++t) {
    const __bf16* si = (t & 1) ? s1 : s0;
    __bf16*       so = (t & 1) ? s0 : s1;
    float* hTp = (t == Tn - 1) ? hT : nullptr;
    k_step<<<48, 128, 0, stream>>>(PRE, bWh, inv, si, so, hidden, hTp, t);
  }

  (void)in_sizes; (void)n_in; (void)out_size; (void)ws_size;
}
